// CustomLSTM_81544249082463
// MI455X (gfx1250) — compile-verified, matrix operand order checked by reading, not run on hardware
//
#include <hip/hip_runtime.h>

typedef __bf16 bf16_t;
typedef __attribute__((ext_vector_type(16))) __bf16 v16bf;
typedef __attribute__((ext_vector_type(8)))  __bf16 v8bf;
typedef __attribute__((ext_vector_type(8)))  float  v8f;
typedef __attribute__((ext_vector_type(4)))  float  v4f;

#define B_   128
#define T_   512
#define E_   512
#define H_   1024
#define G_   4096   // 4*H
#define K_   1536   // E + H

// ---------------------------------------------------------------------------
// Build Wcat[g, 0:512] = W_x{gate}[r, :], Wcat[g, 512:1536] = W_h{gate}[r, :]
// gate order i, f, g, o ; converted fp32 -> bf16. Row-major: row g is exactly
// the contiguous K-stream one WMMA B-matrix lane (output column g) consumes.
// ---------------------------------------------------------------------------
__global__ __launch_bounds__(256) void k_prep(
    const float* __restrict__ Wii, const float* __restrict__ Wif,
    const float* __restrict__ Wig, const float* __restrict__ Wio,
    const float* __restrict__ Whi, const float* __restrict__ Whf,
    const float* __restrict__ Whg, const float* __restrict__ Who,
    bf16_t* __restrict__ Wcat)
{
    const int g    = blockIdx.x;        // 0..4095
    const int gate = g >> 10;
    const int r    = g & (H_ - 1);
    const float* Wx;
    const float* Wh;
    switch (gate) {
        case 0:  Wx = Wii; Wh = Whi; break;
        case 1:  Wx = Wif; Wh = Whf; break;
        case 2:  Wx = Wig; Wh = Whg; break;
        default: Wx = Wio; Wh = Who; break;
    }
    bf16_t* row = Wcat + (size_t)g * K_;
    for (int k = threadIdx.x; k < E_; k += blockDim.x)
        row[k] = (bf16_t)Wx[(size_t)r * E_ + k];
    for (int k = threadIdx.x; k < H_; k += blockDim.x)
        row[E_ + k] = (bf16_t)Wh[(size_t)r * H_ + k];
}

// ---------------------------------------------------------------------------
// Embedding gather + fp32 -> bf16, once for all timesteps (pad row -> zeros).
// One block per (b, t) token; 128 threads x 4 elements = E.
// ---------------------------------------------------------------------------
__global__ __launch_bounds__(128) void k_embed(
    const int* __restrict__ x, const float* __restrict__ emb,
    bf16_t* __restrict__ xe)
{
    const int bt  = blockIdx.x;              // b * T + t
    const int tok = x[bt];
    const float* src = emb + (size_t)tok * E_;
    bf16_t*      dst = xe + (size_t)bt * E_;
    const int k = threadIdx.x * 4;
    if (tok != 0) {
        v4f f = *(const v4f*)(src + k);
        dst[k]     = (bf16_t)f[0];
        dst[k + 1] = (bf16_t)f[1];
        dst[k + 2] = (bf16_t)f[2];
        dst[k + 3] = (bf16_t)f[3];
    } else {
        dst[k] = dst[k + 1] = dst[k + 2] = dst[k + 3] = (bf16_t)0.0f;
    }
}

// ---------------------------------------------------------------------------
// Zero h0 (bf16), c (f32), out_h (f32): 128*1024 elements each.
// ---------------------------------------------------------------------------
__global__ __launch_bounds__(256) void k_init(
    bf16_t* __restrict__ h0, float* __restrict__ c, float* __restrict__ outh)
{
    const int i = blockIdx.x * blockDim.x + threadIdx.x;   // grid covers 131072
    h0[i]   = (bf16_t)0.0f;
    c[i]    = 0.0f;
    outh[i] = 0.0f;
}

// ---------------------------------------------------------------------------
// One LSTM timestep, fully fused:
//   1) async-copy the block's A panel  [xe_t | h_prev] (16 x 1536 bf16, 48 KB)
//      global -> LDS via GLOBAL_LOAD_ASYNC_TO_LDS_B128 (ASYNCcnt)
//   2) z = A x Wcat^T with v_wmma_f32_16x16x32_bf16, A from LDS, W streamed
//      from L2 (12 MB panel, global_load_b128 + global_prefetch_b8)
//   3) gates -> c, h update -> masked output write (LDS exchange)
// Block = 256 threads = 8 waves. Wave w: gate = w>>1, 32-col pair = w&1,
// each wave computes TWO 16x16 N-tiles sharing its A registers.
// Block tile: 16 batch rows x 64 hidden cols x 4 gates. Grid (H/64, B/16).
// ---------------------------------------------------------------------------
__global__ __launch_bounds__(256) void k_step(
    const bf16_t* __restrict__ xe,     // [B, T, E] bf16 (pre-gathered)
    const int*   __restrict__ lengths, // [B]
    const bf16_t* __restrict__ Wcat,   // [G, K]
    const float* __restrict__ b_i, const float* __restrict__ b_f,
    const float* __restrict__ b_g, const float* __restrict__ b_o,
    const bf16_t* __restrict__ h_in,   // [B, H] bf16
    bf16_t*       __restrict__ h_out,  // [B, H] bf16
    float*        __restrict__ c,      // [B, H]
    float*        __restrict__ out_h,  // [B, H]
    int t)
{
    __shared__ __align__(16) bf16_t Ash[16][K_];  // 48 KB: [xe_t | h] panel
    __shared__ float zs[4][16][66];               // 64 cols + pad (bank-safe)

    const int tid   = threadIdx.x;
    const int lane  = tid & 31;
    const int w     = tid >> 5;        // wave 0..7
    const int gate  = w >> 1;
    const int npair = w & 1;           // which 32-col pair of the 64
    const int ln    = lane & 15;
    const int sel   = lane >> 4;       // 0: K[0..7],K[16..23]  1: K[8..15],K[24..31]

    const int m0 = blockIdx.y * 16;    // batch-row base
    const int h0 = blockIdx.x * 64;    // hidden-col base

    // ---- Stage A panel into LDS: 3072 x b128 chunks, 12 per thread --------
    // Row r (3072 B) = xe[m0+r, t, 0:512] (1024 B) ++ h_in[m0+r, 0:1024] (2048 B)
    const uint32_t ldsBase = (uint32_t)(uintptr_t)(void*)&Ash[0][0];
    #pragma unroll
    for (int i = 0; i < 12; ++i) {
        const int li   = tid + i * 256;          // chunk id 0..3071
        const int row  = li / 192;               // 192 chunks per row
        const int boff = (li % 192) * 16;        // byte offset within row
        const char* src = (boff < 1024)
            ? (const char*)(xe + ((size_t)(m0 + row) * T_ + t) * E_) + boff
            : (const char*)(h_in + (size_t)(m0 + row) * H_) + (boff - 1024);
        const uint32_t dst = ldsBase + (uint32_t)(row * 3072 + boff);
        asm volatile("global_load_async_to_lds_b128 %0, %1, off"
                     :: "v"(dst), "v"(src) : "memory");
    }
    asm volatile("s_wait_asynccnt 0x0" ::: "memory");
    __syncthreads();

    // ---- B operand: two Wcat rows (two output columns per wave) ----
    const int     gcol0 = gate * H_ + h0 + npair * 32 + ln;
    const bf16_t* wrow0 = Wcat + (size_t)gcol0 * K_;
    const bf16_t* wrow1 = wrow0 + (size_t)16 * K_;

    v8f acc0 = {};
    v8f acc1 = {};

    // ---- Single unified K loop (A panel and Wcat are both [x | h] concat) --
    for (int kt = 0; kt < K_; kt += 32) {
        const int ka = kt + sel * 8;
        v8bf a0 = *(const v8bf*)&Ash[ln][ka];        // ds_load_b128
        v8bf a1 = *(const v8bf*)&Ash[ln][ka + 16];
        v8bf p0 = *(const v8bf*)(wrow0 + ka);
        v8bf p1 = *(const v8bf*)(wrow0 + ka + 16);
        v8bf q0 = *(const v8bf*)(wrow1 + ka);
        v8bf q1 = *(const v8bf*)(wrow1 + ka + 16);
        __builtin_prefetch(wrow0 + ka + 64, 0, 3);   // global_prefetch_b8
        __builtin_prefetch(wrow1 + ka + 64, 0, 3);
        v16bf a  = __builtin_shufflevector(a0, a1, 0,1,2,3,4,5,6,7,8,9,10,11,12,13,14,15);
        v16bf bp = __builtin_shufflevector(p0, p1, 0,1,2,3,4,5,6,7,8,9,10,11,12,13,14,15);
        v16bf bq = __builtin_shufflevector(q0, q1, 0,1,2,3,4,5,6,7,8,9,10,11,12,13,14,15);
        acc0 = __builtin_amdgcn_wmma_f32_16x16x32_bf16(
                   false, a, false, bp, (short)0, acc0, false, false);
        acc1 = __builtin_amdgcn_wmma_f32_16x16x32_bf16(
                   false, a, false, bq, (short)0, acc1, false, false);
    }

    // C/D layout: VGPR j, lane L -> row m = j + 8*(L/16), col n = L%16
    #pragma unroll
    for (int j = 0; j < 8; ++j) {
        zs[gate][j + 8 * sel][npair * 32 + ln]      = acc0[j];
        zs[gate][j + 8 * sel][npair * 32 + 16 + ln] = acc1[j];
    }

    __syncthreads();

    // Gate nonlinearities + state update: 16x64 = 1024 elems, 4 per thread
    for (int e = tid; e < 1024; e += 256) {
        const int r  = e >> 6;
        const int hc = e & 63;
        const int b  = m0 + r;
        const int h  = h0 + hc;
        float zi = zs[0][r][hc] + b_i[h];
        float zf = zs[1][r][hc] + b_f[h];
        float zg = zs[2][r][hc] + b_g[h];
        float zo = zs[3][r][hc] + b_o[h];
        zi = 1.0f / (1.0f + __expf(-zi));
        zf = 1.0f / (1.0f + __expf(-zf));
        zg = tanhf(zg);
        zo = 1.0f / (1.0f + __expf(-zo));
        const size_t idx = (size_t)b * H_ + h;
        const float cn = zf * c[idx] + zi * zg;
        const float hn = zo * tanhf(cn);
        c[idx]     = cn;
        h_out[idx] = (bf16_t)hn;
        if (lengths[b] == t + 1) out_h[idx] = hn;
    }
}

// ---------------------------------------------------------------------------
// Final FC: out[b] = dot(out_h[b,:], fc_w) + fc_b   (O = 1)
// ---------------------------------------------------------------------------
__global__ __launch_bounds__(256) void k_fc(
    const float* __restrict__ out_h, const float* __restrict__ fc_w,
    const float* __restrict__ fc_b, float* __restrict__ out)
{
    __shared__ float red[256];
    const int b = blockIdx.x;
    float s = 0.0f;
    for (int h = threadIdx.x; h < H_; h += 256)
        s += out_h[(size_t)b * H_ + h] * fc_w[h];
    red[threadIdx.x] = s;
    __syncthreads();
    for (int off = 128; off > 0; off >>= 1) {
        if (threadIdx.x < off) red[threadIdx.x] += red[threadIdx.x + off];
        __syncthreads();
    }
    if (threadIdx.x == 0) out[b] = red[0] + fc_b[0];
}

// ---------------------------------------------------------------------------
extern "C" void kernel_launch(void* const* d_in, const int* in_sizes, int n_in,
                              void* d_out, int out_size, void* d_ws, size_t ws_size,
                              hipStream_t stream)
{
    const int*   x       = (const int*)  d_in[0];
    const int*   lengths = (const int*)  d_in[1];
    const float* emb     = (const float*)d_in[2];
    const float* Wii = (const float*)d_in[3];
    const float* Whi = (const float*)d_in[4];
    const float* bi  = (const float*)d_in[5];
    const float* Wif = (const float*)d_in[6];
    const float* Whf = (const float*)d_in[7];
    const float* bf  = (const float*)d_in[8];
    const float* Wig = (const float*)d_in[9];
    const float* Whg = (const float*)d_in[10];
    const float* bg  = (const float*)d_in[11];
    const float* Wio = (const float*)d_in[12];
    const float* Who = (const float*)d_in[13];
    const float* bo  = (const float*)d_in[14];
    const float* fcw = (const float*)d_in[15];
    const float* fcb = (const float*)d_in[16];

    char* ws = (char*)d_ws;
    size_t off = 0;
    bf16_t* Wcat  = (bf16_t*)(ws + off); off += (size_t)G_ * K_ * sizeof(bf16_t);       // 12 MB
    bf16_t* xebuf = (bf16_t*)(ws + off); off += (size_t)B_ * T_ * E_ * sizeof(bf16_t);  // 64 MB
    bf16_t* hbuf0 = (bf16_t*)(ws + off); off += (size_t)B_ * H_ * sizeof(bf16_t);
    bf16_t* hbuf1 = (bf16_t*)(ws + off); off += (size_t)B_ * H_ * sizeof(bf16_t);
    float*  cbuf  = (float*) (ws + off); off += (size_t)B_ * H_ * sizeof(float);
    float*  outh  = (float*) (ws + off); off += (size_t)B_ * H_ * sizeof(float);

    k_prep<<<G_, 256, 0, stream>>>(Wii, Wif, Wig, Wio, Whi, Whf, Whg, Who, Wcat);
    k_embed<<<B_ * T_, 128, 0, stream>>>(x, emb, xebuf);
    k_init<<<(B_ * H_) / 256, 256, 0, stream>>>(hbuf0, cbuf, outh);

    for (int t = 0; t < T_; ++t) {
        bf16_t* hin  = (t & 1) ? hbuf1 : hbuf0;
        bf16_t* hout = (t & 1) ? hbuf0 : hbuf1;
        k_step<<<dim3(H_ / 64, B_ / 16), 256, 0, stream>>>(
            xebuf, lengths, Wcat, bi, bf, bg, bo, hin, hout, cbuf, outh, t);
    }

    k_fc<<<B_, 256, 0, stream>>>(outh, fcw, fcb, (float*)d_out);
}
